// dCGF_ES_65386582114980
// MI455X (gfx1250) — matrix operand order repaired and checked by reading, not compiled, not run on hardware
//
#include <hip/hip_runtime.h>
#include <hip/hip_bf16.h>

// Problem constants (from reference)
#define N_GF      4096
#define TOTAL_COLS 65536
#define BSEG      2048
#define ENC_H     200
#define ENC_HP    208   // padded to 13 * 16 for WMMA tiles
#define Z_DIM     30
#define DEC_H     100

typedef __attribute__((ext_vector_type(16))) __bf16 v16bf;
typedef __attribute__((ext_vector_type(8)))  float  v8f;

static __device__ __forceinline__ unsigned short f2bf_bits(float f) {
    unsigned u = __float_as_uint(f);
    u += 0x7FFFu + ((u >> 16) & 1u);   // round-to-nearest-even
    return (unsigned short)(u >> 16);
}
static __device__ __forceinline__ __bf16 bits2bf(unsigned short s) {
    union { unsigned short u; __bf16 h; } v; v.u = s; return v.h;
}

// ---------------------------------------------------------------------------
// Kernel A: segment boundaries via lower_bound on sorted segment_ids.
// seg_start[s] = first row index with id >= s; seg_start[BSEG] = TOTAL_COLS.
// ---------------------------------------------------------------------------
__global__ void k_seg_bounds(const int* __restrict__ seg, int* __restrict__ start) {
    int s = blockIdx.x * blockDim.x + threadIdx.x;
    if (s > BSEG) return;
    int lo = 0, hi = TOTAL_COLS;
    while (lo < hi) {
        int mid = (lo + hi) >> 1;
        if (seg[mid] < s) lo = mid + 1; else hi = mid;
    }
    start[s] = lo;
}

// ---------------------------------------------------------------------------
// Kernel B: streaming segment sum (the 1 GiB HBM-bound pass, ~46 us roofline).
// Grid: (4 col-chunks, 2048 segments). Each thread owns 4 columns (float4),
// coalesced b128 loads across the block, accumulate f32, store bf16.
// Row loads are independent of the accumulate chain; unroll to keep several
// b128 loads in flight per thread.
// ---------------------------------------------------------------------------
__global__ void k_segsum_bf16(const float4* __restrict__ u4,
                              const int* __restrict__ start,
                              unsigned short* __restrict__ w) {
    int seg = blockIdx.y;
    int c4  = blockIdx.x * blockDim.x + threadIdx.x;   // 0..1023 (float4 index)
    int r0 = start[seg], r1 = start[seg + 1];
    float4 acc = make_float4(0.f, 0.f, 0.f, 0.f);
    int r = r0;
#pragma unroll 1
    for (; r + 4 <= r1; r += 4) {
        float4 v0 = u4[(size_t)(r + 0) * (N_GF / 4) + c4];
        float4 v1 = u4[(size_t)(r + 1) * (N_GF / 4) + c4];
        float4 v2 = u4[(size_t)(r + 2) * (N_GF / 4) + c4];
        float4 v3 = u4[(size_t)(r + 3) * (N_GF / 4) + c4];
        acc.x += v0.x + v1.x + v2.x + v3.x;
        acc.y += v0.y + v1.y + v2.y + v3.y;
        acc.z += v0.z + v1.z + v2.z + v3.z;
        acc.w += v0.w + v1.w + v2.w + v3.w;
    }
    for (; r < r1; ++r) {
        float4 v = u4[(size_t)r * (N_GF / 4) + c4];
        acc.x += v.x; acc.y += v.y; acc.z += v.z; acc.w += v.w;
    }
    ushort4 o;
    o.x = f2bf_bits(acc.x); o.y = f2bf_bits(acc.y);
    o.z = f2bf_bits(acc.z); o.w = f2bf_bits(acc.w);
    ((ushort4*)w)[(size_t)seg * (N_GF / 4) + c4] = o;
}

// ---------------------------------------------------------------------------
// Kernel: convert W1 (4096 x 200 f32, row-major K x N) -> bf16 TRANSPOSED to
// (208 x 4096, n-major), zero-padded cols 200..207. The transpose makes each
// lane's WMMA B-fragment (16 K-contiguous bf16 for one column n) two
// contiguous b128 loads instead of 16 strided u16 gathers.
// ---------------------------------------------------------------------------
__global__ void k_w1t_bf16(const float* __restrict__ W1, unsigned short* __restrict__ w1t) {
    int idx = blockIdx.x * blockDim.x + threadIdx.x;
    if (idx >= ENC_HP * N_GF) return;
    int n = idx / N_GF, k = idx % N_GF;
    float v = (n < ENC_H) ? W1[k * ENC_H + n] : 0.f;
    w1t[idx] = f2bf_bits(v);
}

// ---------------------------------------------------------------------------
// Kernel C: h = relu(w @ W1 + b1) via v_wmma_f32_16x16x32_bf16.
// One wave computes one 16x16 output tile; 1664 waves -> 208 blocks so the
// (L2-load-bound) GEMM spreads across ~208 WGPs. All 8 waves of a block share
// the same N-tile, so B fragments hit WGP$.
// Fragment layouts per CDNA5 ISA 7.12.2 (16-bit A 16x32 / B 32x16):
//   A: lane holds row m = lane&15; K = k0 + 8*(lane>>4) + i      (i<8)
//                                  K = k0 + 16 + 8*(lane>>4) + i-8 (i>=8)
//      -> two contiguous 16B loads per lane per K-step (row-major w_bf16).
//   B: lane holds col n = nTile*16 + (lane&15); K same as A (table order)
//      -> two contiguous 16B loads per lane per K-step (n-major w1t).
//   C/D: VGPR j -> row j + 8*(lane>>4), col lane&15.
// ---------------------------------------------------------------------------
__global__ void k_gemm1_wmma(const unsigned short* __restrict__ w,
                             const unsigned short* __restrict__ w1t,
                             const float* __restrict__ b1,
                             float* __restrict__ h) {
    const int NT = ENC_HP / 16;                 // 13 N-tiles
    int nTile  = blockIdx.x % NT;               // same N-tile for whole block
    int mTile  = (blockIdx.x / NT) * 8 + (threadIdx.x >> 5);
    int lane   = threadIdx.x & 31;
    int laneLo = lane & 15;
    int laneHi = lane >> 4;
    int n = nTile * 16 + laneLo;

    v8f c = 0.0f;

    const unsigned short* arow = w   + (size_t)(mTile * 16 + laneLo) * N_GF + 8 * laneHi;
    const unsigned short* bcol = w1t + (size_t)n * N_GF + 8 * laneHi;

#pragma unroll 2
    for (int k0 = 0; k0 < N_GF; k0 += 32) {
        union { uint4 q; unsigned short s[8]; } a0, a1, bq0, bq1;
        a0.q  = *(const uint4*)(arow + k0);        // K = k0 + 8*laneHi + 0..7
        a1.q  = *(const uint4*)(arow + k0 + 16);   // K = k0 + 16 + 8*laneHi + 0..7
        bq0.q = *(const uint4*)(bcol + k0);        // K = k0 + 8*laneHi + 0..7
        bq1.q = *(const uint4*)(bcol + k0 + 16);   // K = k0 + 16 + 8*laneHi + 0..7
        v16bf afrag, bfrag;
#pragma unroll
        for (int i = 0; i < 8; ++i) {
            afrag[i]     = bits2bf(a0.s[i]);
            afrag[i + 8] = bits2bf(a1.s[i]);
            bfrag[i]     = bits2bf(bq0.s[i]);
            bfrag[i + 8] = bits2bf(bq1.s[i]);
        }
        c = __builtin_amdgcn_wmma_f32_16x16x32_bf16(
            false, afrag, false, bfrag, (short)0, c, false, false);
    }

    if (n < ENC_H) {
        float bias = b1[n];
        int rowBase = mTile * 16 + 8 * laneHi;
#pragma unroll
        for (int j = 0; j < 8; ++j) {
            float v = c[j] + bias;
            h[(size_t)(rowBase + j) * ENC_H + n] = v > 0.f ? v : 0.f;
        }
    }
}

// ---------------------------------------------------------------------------
// Kernel D: tail MLP, one wave32 per row. ~18 KFLOP/row -> negligible.
// ---------------------------------------------------------------------------
__global__ void k_tail(const float* __restrict__ h,
                       const float* __restrict__ W2, const float* __restrict__ b2,
                       const float* __restrict__ W3, const float* __restrict__ b3,
                       const float* __restrict__ W4, const float* __restrict__ b4,
                       float* __restrict__ y) {
    __shared__ float hs[ENC_H];
    __shared__ float zs[Z_DIM];
    __shared__ float ds_[DEC_H];
    __shared__ float red[32];
    int row  = blockIdx.x;
    int lane = threadIdx.x;   // blockDim.x == 32 (one wave)

    for (int k = lane; k < ENC_H; k += 32) hs[k] = h[(size_t)row * ENC_H + k];
    __syncthreads();

    if (lane < Z_DIM) {
        float acc = b2[lane];
        for (int k = 0; k < ENC_H; ++k) acc += hs[k] * W2[k * Z_DIM + lane];
        zs[lane] = acc > 0.f ? acc : 0.f;
    }
    __syncthreads();

    for (int j = lane; j < DEC_H; j += 32) {
        float acc = b3[j];
        for (int k = 0; k < Z_DIM; ++k) acc += zs[k] * W3[k * DEC_H + j];
        ds_[j] = acc > 0.f ? acc : 0.f;
    }
    __syncthreads();

    float p = 0.f;
    for (int j = lane; j < DEC_H; j += 32) p += ds_[j] * W4[j];
    red[lane] = p;
    __syncthreads();
    if (lane == 0) {
        float acc = b4[0];
        for (int i = 0; i < 32; ++i) acc += red[i];
        y[row] = acc;
    }
}

// ---------------------------------------------------------------------------
extern "C" void kernel_launch(void* const* d_in, const int* in_sizes, int n_in,
                              void* d_out, int out_size, void* d_ws, size_t ws_size,
                              hipStream_t stream) {
    (void)in_sizes; (void)n_in; (void)out_size; (void)ws_size;
    const float* u_flat = (const float*)d_in[0];
    const float* W1     = (const float*)d_in[1];
    const float* b1     = (const float*)d_in[2];
    const float* W2     = (const float*)d_in[3];
    const float* b2     = (const float*)d_in[4];
    const float* W3     = (const float*)d_in[5];
    const float* b3     = (const float*)d_in[6];
    const float* W4     = (const float*)d_in[7];
    const float* b4     = (const float*)d_in[8];
    const int*   segids = (const int*)d_in[9];
    float* y = (float*)d_out;

    // Workspace layout (all offsets 16B-aligned):
    //   [0, 16384)        seg_start  (BSEG+1 ints)
    //   [16384, +16 MiB)  w_bf16     (2048 x 4096 bf16, row-major)
    //   [.., +1.66 MiB)   W1^T bf16  (208 x 4096 bf16, n-major, zero-padded)
    //   [.., +1.56 MiB)   h          (2048 x 200 f32)
    char* ws = (char*)d_ws;
    int*            seg_start = (int*)ws;
    unsigned short* w_bf      = (unsigned short*)(ws + 16384);
    unsigned short* w1t       = (unsigned short*)(ws + 16384 + (size_t)BSEG * N_GF * 2);
    float*          hbuf      = (float*)(ws + 16384 + (size_t)BSEG * N_GF * 2
                                                    + (size_t)ENC_HP * N_GF * 2);

    k_seg_bounds<<<(BSEG + 1 + 255) / 256, 256, 0, stream>>>(segids, seg_start);

    k_segsum_bf16<<<dim3(N_GF / (256 * 4), BSEG), 256, 0, stream>>>(
        (const float4*)u_flat, seg_start, w_bf);

    k_w1t_bf16<<<(ENC_HP * N_GF + 255) / 256, 256, 0, stream>>>(W1, w1t);

    // 13 N-tiles * 16 m-groups = 208 blocks, 8 waves each (one 16x16 tile/wave)
    k_gemm1_wmma<<<(ENC_HP / 16) * (BSEG / 16 / 8), 256, 0, stream>>>(
        w_bf, w1t, b1, hbuf);

    k_tail<<<BSEG, 32, 0, stream>>>(hbuf, W2, b2, W3, b3, W4, b4, y);
}
